// CoreNNModel_77833397338719
// MI455X (gfx1250) — matrix operand order repaired and checked by reading, not compiled, not run on hardware
//
#include <hip/hip_runtime.h>
#include <stdint.h>
#include <stddef.h>

// ---------------- model constants ----------------
#define MB 4        // batch
#define MS 512      // sequence
#define MD 1024     // embedding dim
#define MV 65536    // vocab
#define MNS 512     // memory slots
#define MT (MB*MS)  // 2048 token rows; row index = s*4 + b

// ---------------- CDNA5 types ----------------
typedef __attribute__((ext_vector_type(16))) __bf16        v16bf;
typedef __attribute__((ext_vector_type(8)))  float         v8f;
typedef __attribute__((ext_vector_type(4)))  unsigned int  u32x4;
typedef __attribute__((ext_vector_type(4)))  int           i32x4;
typedef __attribute__((ext_vector_type(8)))  int           i32x8;

// Tensor Data Mover availability (arity differs across toolchains)
#if defined(__has_builtin)
#  if __has_builtin(__builtin_amdgcn_tensor_load_to_lds)
#    define HAVE_TDM 1
#  endif
#endif
#ifndef HAVE_TDM
#  define HAVE_TDM 0
#endif

union FragBF { v16bf v; u32x4 q[2]; };

__device__ __forceinline__ unsigned short f2bf(float f) {
  unsigned int u = __builtin_bit_cast(unsigned int, f);
  u += 0x7FFFu + ((u >> 16) & 1u);          // round-to-nearest-even
  return (unsigned short)(u >> 16);
}

__device__ __forceinline__ v8f wmma_bf16(FragBF a, FragBF b, v8f c) {
  // D = A(16x32 bf16) * B(32x16 bf16) + C(16x16 f32)
  return __builtin_amdgcn_wmma_f32_16x16x32_bf16(
      false, a.v, false, b.v, (short)0, c, false, false);
}

__device__ __forceinline__ float sigmoidf_(float x) { return 1.0f / (1.0f + expf(-x)); }

// LDS byte offset of a __shared__ object: low 32 bits of the flat address
// (LDS aperture maps addr[31:0] directly to the LDS offset, ISA 10.2).
__device__ __forceinline__ unsigned lds_off(const void* p) {
  return (unsigned)(uintptr_t)p;
}

#if HAVE_TDM
// ---------------------------------------------------------------------
// TDM 2D tile load: rows x 64B tile (16 dwords/row), row stride/tensor
// width in dword units. D# per CDNA5 ISA.
// Issued once per wave (EXEC ignored); tracked by TENSORcnt.
// ---------------------------------------------------------------------
__device__ __forceinline__ void tdm_load_tile(unsigned lds_addr, const void* gptr,
                                              int stride_dw, int rows) {
  unsigned long long ga = (unsigned long long)(uintptr_t)gptr;
  u32x4 g0;
  g0[0] = 1u;                                        // count=1, user descriptor
  g0[1] = lds_addr;                                  // LDS byte address
  g0[2] = (unsigned)ga;                              // global addr [31:0]
  g0[3] = (unsigned)((ga >> 32) & 0x1FFFFFFu)        // global addr [56:32]
          | (2u << 30);                              // type=2 ("image")
  i32x8 g1;
  g1[0] = (int)(2u << 16);                           // data_size=4B, wg_mask=0
  g1[1] = (int)(((unsigned)stride_dw & 0xFFFFu) << 16);        // tensor_dim0 lo16
  g1[2] = (int)(((unsigned)stride_dw >> 16) & 0xFFFFu)         // tensor_dim0 hi16
          | (int)(((unsigned)rows & 0xFFFFu) << 16);           // tensor_dim1 lo16
  g1[3] = (int)(16u << 16);                          // tensor_dim1 hi=0, tile_dim0=16 dw
  g1[4] = (int)((unsigned)rows & 0xFFFFu);           // tile_dim1=rows, tile_dim2=0
  g1[5] = (int)stride_dw;                            // tensor_dim0_stride lo32
  g1[6] = 0;                                         // stride hi16, dim1_stride lo16
  g1[7] = 0;
  i32x4 gz = {0, 0, 0, 0};
#if defined(__clang_major__) && (__clang_major__ >= 23)
  i32x8 gz8 = {0, 0, 0, 0, 0, 0, 0, 0};
  __builtin_amdgcn_tensor_load_to_lds(g0, g1, gz, gz, gz8, 0);
#else
  __builtin_amdgcn_tensor_load_to_lds(g0, g1, gz, gz, 0);
#endif
}
#endif  // HAVE_TDM

// =====================================================================
// 1) Embedding gather + input LayerNorm -> x_bf [T, D] bf16 (row = s*4+b)
// =====================================================================
__global__ __launch_bounds__(256) void embed_ln_kernel(
    const int* __restrict__ ids, const float* __restrict__ tok,
    const float* __restrict__ pos, const float* __restrict__ g,
    const float* __restrict__ bta, unsigned short* __restrict__ xbf) {
  const int t = blockIdx.x;           // token row: s*4+b
  const int b = t & 3, s = t >> 2;
  const int id = ids[b * MS + s];
  const int tid = threadIdx.x;
  __shared__ float r1[256], r2[256];
  float vals[4]; float lsum = 0.f, lsq = 0.f;
#pragma unroll
  for (int j = 0; j < 4; ++j) {
    int d = tid + j * 256;
    float v = tok[(size_t)id * MD + d] + pos[(size_t)s * MD + d];
    vals[j] = v; lsum += v; lsq += v * v;
  }
  r1[tid] = lsum; r2[tid] = lsq; __syncthreads();
  for (int off = 128; off > 0; off >>= 1) {
    if (tid < off) { r1[tid] += r1[tid + off]; r2[tid] += r2[tid + off]; }
    __syncthreads();
  }
  const float mu = r1[0] / MD;
  const float var = r2[0] / MD - mu * mu;
  const float rstd = rsqrtf(var + 1e-5f);
#pragma unroll
  for (int j = 0; j < 4; ++j) {
    int d = tid + j * 256;
    float v = (vals[j] - mu) * rstd * g[d] + bta[d];
    xbf[(size_t)t * MD + d] = f2bf(v);
  }
}

// =====================================================================
// 2) Output LayerNorm: igpm f32 [T,D] -> final_bf [T,D]
// =====================================================================
__global__ __launch_bounds__(256) void ln_out_kernel(
    const float* __restrict__ in, const float* __restrict__ g,
    const float* __restrict__ bta, unsigned short* __restrict__ obf) {
  const int t = blockIdx.x;
  const int tid = threadIdx.x;
  __shared__ float r1[256], r2[256];
  float vals[4]; float lsum = 0.f, lsq = 0.f;
#pragma unroll
  for (int j = 0; j < 4; ++j) {
    int d = tid + j * 256;
    float v = in[(size_t)t * MD + d];
    vals[j] = v; lsum += v; lsq += v * v;
  }
  r1[tid] = lsum; r2[tid] = lsq; __syncthreads();
  for (int off = 128; off > 0; off >>= 1) {
    if (tid < off) { r1[tid] += r1[tid + off]; r2[tid] += r2[tid + off]; }
    __syncthreads();
  }
  const float mu = r1[0] / MD;
  const float rstd = rsqrtf(r2[0] / MD - mu * mu + 1e-5f);
#pragma unroll
  for (int j = 0; j < 4; ++j) {
    int d = tid + j * 256;
    obf[(size_t)t * MD + d] = f2bf((vals[j] - mu) * rstd * g[d] + bta[d]);
  }
}

// =====================================================================
// 3) f32 -> bf16 convert (same layout)
// =====================================================================
__global__ __launch_bounds__(256) void convert_bf16_kernel(
    const float* __restrict__ in, unsigned short* __restrict__ out, int n) {
  int i = blockIdx.x * 256 + threadIdx.x;
  if (i < n) out[i] = f2bf(in[i]);
}

// =====================================================================
// 4) f32 [R,C] -> bf16 [C,R] transpose-convert (R,C multiples of 32)
// =====================================================================
__global__ __launch_bounds__(256) void transpose_cvt_kernel(
    const float* __restrict__ in, unsigned short* __restrict__ out, int R, int C) {
  __shared__ float tile[32][33];
  const int bx = blockIdx.x * 32;   // C dim
  const int by = blockIdx.y * 32;   // R dim
  const int tx = threadIdx.x & 31;
  const int ty = threadIdx.x >> 5;  // 0..7
#pragma unroll
  for (int j = 0; j < 4; ++j)
    tile[ty + j * 8][tx] = in[(size_t)(by + ty + j * 8) * C + bx + tx];
  __syncthreads();
#pragma unroll
  for (int j = 0; j < 4; ++j)
    out[(size_t)(bx + ty + j * 8) * R + by + tx] = f2bf(tile[tx][ty + j * 8]);
}

// =====================================================================
// 5) Generic bf16 WMMA GEMM:  C[M,N](f32) = A[M,K](bf16) * Bt[N,K]^T (+bias)(+addsrc)
//    Block tile 128x128, BK=32, 8 waves, each wave = 4x2 WMMA tiles.
//    TDM path: wave0/wave1 DMA the A/B tiles to LDS (double-buffered),
//    tracked by TENSORcnt; the other waves only compute.
//    mode==1: logits row-permute store: row=s*4+b -> out[(b*MS+s)*N + col]
// =====================================================================
#define BM 128
#define BN 128
#define BK 32

__global__ __launch_bounds__(256) void gemm_bf16_kernel(
    const unsigned short* __restrict__ A,   // [M,K]
    const unsigned short* __restrict__ Bt,  // [N,K]
    const float* __restrict__ bias,         // [N] or nullptr
    const float* __restrict__ addsrc,       // [M,N] or nullptr
    float* __restrict__ C, int M, int N, int K, int mode) {
#if HAVE_TDM
  __shared__ __align__(16) unsigned short As[2][BM * BK];
  __shared__ __align__(16) unsigned short Bs[2][BN * BK];
#else
  __shared__ __align__(16) unsigned short As[1][BM * BK];
  __shared__ __align__(16) unsigned short Bs[1][BN * BK];
#endif

  const int tid  = threadIdx.x;
  const int lane = tid & 31;
  const int wave = tid >> 5;            // 0..7
  const int waveM = (wave & 1) * 64;
  const int waveN = (wave >> 1) * 32;
  const int bm = blockIdx.y * BM;
  const int bn = blockIdx.x * BN;

  const v8f z8 = {0.f,0.f,0.f,0.f,0.f,0.f,0.f,0.f};
  v8f acc[4][2];
#pragma unroll
  for (int i = 0; i < 4; ++i)
#pragma unroll
    for (int j = 0; j < 2; ++j) acc[i][j] = z8;

  const int c0  = (lane >> 4) * 8;      // A fragment K chunk base {0,8}
  const int bk0 = (lane >> 4) * 8;      // B fragment K chunk base {0,8}
  const int l15 = lane & 15;

#if HAVE_TDM
  // ---------------- TDM double-buffered pipeline ----------------
  const int niter = K / BK;
  const int Kdw = K >> 1;               // row stride in dwords
  const unsigned short* gA = A  + (size_t)bm * K;
  const unsigned short* gB = Bt + (size_t)bn * K;

  if (wave == 0) tdm_load_tile(lds_off(&As[0][0]), gA, Kdw, BM);
  if (wave == 1) tdm_load_tile(lds_off(&Bs[0][0]), gB, Kdw, BN);

  for (int i = 0; i < niter; ++i) {
    const int cur = i & 1;
    if (i + 1 < niter) {
      if (wave == 0) tdm_load_tile(lds_off(&As[cur ^ 1][0]), gA + (size_t)(i + 1) * BK, Kdw, BM);
      if (wave == 1) tdm_load_tile(lds_off(&Bs[cur ^ 1][0]), gB + (size_t)(i + 1) * BK, Kdw, BN);
      if (wave < 2) __builtin_amdgcn_s_wait_tensorcnt((short)1);
    } else {
      if (wave < 2) __builtin_amdgcn_s_wait_tensorcnt((short)0);
    }
    __syncthreads();                    // tile 'cur' valid in LDS for all waves

    FragBF afr[4], bfr[2];
#pragma unroll
    for (int mt = 0; mt < 4; ++mt) {
      int r = waveM + mt * 16 + l15;
      afr[mt].q[0] = *(const u32x4*)&As[cur][r * BK + c0];
      afr[mt].q[1] = *(const u32x4*)&As[cur][r * BK + c0 + 16];
    }
#pragma unroll
    for (int nt = 0; nt < 2; ++nt) {
      int n = waveN + nt * 16 + l15;
      bfr[nt].q[0] = *(const u32x4*)&Bs[cur][n * BK + bk0];
      bfr[nt].q[1] = *(const u32x4*)&Bs[cur][n * BK + bk0 + 16];
    }
#pragma unroll
    for (int mt = 0; mt < 4; ++mt)
#pragma unroll
      for (int nt = 0; nt < 2; ++nt)
        acc[mt][nt] = wmma_bf16(afr[mt], bfr[nt], acc[mt][nt]);
    __syncthreads();                    // done reading 'cur' before it is re-filled
  }
#else
  // ---------------- fallback: vector-load staging ----------------
  const int lrow = tid >> 1;            // 0..127
  const int lk   = (tid & 1) * 16;      // 0 or 16
  const unsigned short* pA = A  + (size_t)(bm + lrow) * K + lk;
  const unsigned short* pB = Bt + (size_t)(bn + lrow) * K + lk;

  for (int k0 = 0; k0 < K; k0 += BK) {
    u32x4 a0 = *(const u32x4*)(pA + k0);
    u32x4 b0 = *(const u32x4*)(pB + k0);
    u32x4 a1 = *(const u32x4*)(pA + k0 + 8);
    u32x4 b1 = *(const u32x4*)(pB + k0 + 8);
    *(u32x4*)&As[0][lrow * BK + lk]     = a0;
    *(u32x4*)&As[0][lrow * BK + lk + 8] = a1;
    *(u32x4*)&Bs[0][lrow * BK + lk]     = b0;
    *(u32x4*)&Bs[0][lrow * BK + lk + 8] = b1;
    __syncthreads();

    if (k0 + BK < K) {                  // global_prefetch of next tiles (near cache)
      __builtin_prefetch(pA + k0 + BK, 0, 3);
      __builtin_prefetch(pB + k0 + BK, 0, 3);
    }

    FragBF afr[4], bfr[2];
#pragma unroll
    for (int mt = 0; mt < 4; ++mt) {
      int r = waveM + mt * 16 + l15;
      afr[mt].q[0] = *(const u32x4*)&As[0][r * BK + c0];
      afr[mt].q[1] = *(const u32x4*)&As[0][r * BK + c0 + 16];
    }
#pragma unroll
    for (int nt = 0; nt < 2; ++nt) {
      int n = waveN + nt * 16 + l15;
      bfr[nt].q[0] = *(const u32x4*)&Bs[0][n * BK + bk0];
      bfr[nt].q[1] = *(const u32x4*)&Bs[0][n * BK + bk0 + 16];
    }
#pragma unroll
    for (int mt = 0; mt < 4; ++mt)
#pragma unroll
      for (int nt = 0; nt < 2; ++nt)
        acc[mt][nt] = wmma_bf16(afr[mt], bfr[nt], acc[mt][nt]);
    __syncthreads();
  }
#endif

  // epilogue (C/D layout: lane<16 -> M 0..7, lane>=16 -> M 8..15; N = lane&15)
#pragma unroll
  for (int mt = 0; mt < 4; ++mt) {
#pragma unroll
    for (int nt = 0; nt < 2; ++nt) {
      const int col   = bn + waveN + nt * 16 + l15;
      const int rbase = bm + waveM + mt * 16 + ((lane >> 4) ? 8 : 0);
      const float bv  = bias ? bias[col] : 0.f;
#pragma unroll
      for (int r = 0; r < 8; ++r) {
        const int row = rbase + r;
        float v = acc[mt][nt][r] + bv;
        if (addsrc) v += addsrc[(size_t)row * N + col];
        if (mode == 0) {
          C[(size_t)row * N + col] = v;
        } else {
          const int bb = row & 3, ss = row >> 2;
          C[((size_t)bb * MS + ss) * (size_t)N + col] = v;
        }
      }
    }
  }
}

// =====================================================================
// 6) GRU per-timestep kernels. Hidden state padded to 16 rows (rows 4..15
//    stay zero) so WMMA A fragments need no predication. Weights are
//    L2-resident bf16 [N,K]; fragments stream directly from global.
// =====================================================================
__global__ __launch_bounds__(128) void gru_zr_kernel(
    const unsigned short* __restrict__ hbf,   // [16, D] bf16 (padded)
    const float* __restrict__ hcur,           // [4, D] f32
    const unsigned short* __restrict__ Wzh_t, // [D(n), D(k)] bf16
    const unsigned short* __restrict__ Wrh_t, // [D(n), D(k)] bf16
    const float* __restrict__ Xz,             // [T, D] precomputed x@Wzx + bz
    const float* __restrict__ Xr,
    float* __restrict__ zbuf,                 // [4, D]
    unsigned short* __restrict__ rhbf,        // [16, D] bf16 (padded)
    int t) {
  const int lane = threadIdx.x & 31;
  const int wave = threadIdx.x >> 5;
  const int n0 = blockIdx.x * 64 + wave * 16;
  const unsigned short* Wt = blockIdx.y ? Wrh_t : Wzh_t;
  const float* X = blockIdx.y ? Xr : Xz;

  const v8f z8 = {0.f,0.f,0.f,0.f,0.f,0.f,0.f,0.f};
  v8f acc = z8;
  const int l15 = lane & 15;
  const int c0  = (lane >> 4) * 8;
  const int bk0 = (lane >> 4) * 8;
  const unsigned short* pa = hbf + (size_t)l15 * MD;
  const unsigned short* pb = Wt + (size_t)(n0 + l15) * MD;

  for (int k0 = 0; k0 < MD; k0 += 32) {
    FragBF a, b;
    a.q[0] = *(const u32x4*)(pa + k0 + c0);
    a.q[1] = *(const u32x4*)(pa + k0 + c0 + 16);
    b.q[0] = *(const u32x4*)(pb + k0 + bk0);
    b.q[1] = *(const u32x4*)(pb + k0 + bk0 + 16);
    acc = wmma_bf16(a, b, acc);
  }
  const int col = n0 + l15;
  const int mbase = (lane >> 4) * 8;
#pragma unroll
  for (int r = 0; r < 8; ++r) {
    const int m = mbase + r;
    if (m < MB) {
      const float v = acc[r] + X[(size_t)(t * 4 + m) * MD + col];
      const float gate = sigmoidf_(v);
      if (blockIdx.y == 0) zbuf[m * MD + col] = gate;
      else rhbf[m * MD + col] = f2bf(gate * hcur[m * MD + col]);
    }
  }
}

__global__ __launch_bounds__(128) void gru_h_kernel(
    const unsigned short* __restrict__ rhbf,  // [16, D] bf16
    const unsigned short* __restrict__ Whh_t, // [D, D] bf16
    const float* __restrict__ Xh,             // [T, D]
    const float* __restrict__ zbuf,           // [4, D]
    float* __restrict__ hcur,                 // [4, D]  (in/out)
    unsigned short* __restrict__ hbf,         // [16, D] bf16 (out rows 0..3)
    float* __restrict__ h_all,                // [T, D] f32
    unsigned short* __restrict__ hbf_all,     // [T, D] bf16
    int t) {
  const int lane = threadIdx.x & 31;
  const int wave = threadIdx.x >> 5;
  const int n0 = blockIdx.x * 64 + wave * 16;

  const v8f z8 = {0.f,0.f,0.f,0.f,0.f,0.f,0.f,0.f};
  v8f acc = z8;
  const int l15 = lane & 15;
  const int c0  = (lane >> 4) * 8;
  const int bk0 = (lane >> 4) * 8;
  const unsigned short* pa = rhbf + (size_t)l15 * MD;
  const unsigned short* pb = Whh_t + (size_t)(n0 + l15) * MD;

  for (int k0 = 0; k0 < MD; k0 += 32) {
    FragBF a, b;
    a.q[0] = *(const u32x4*)(pa + k0 + c0);
    a.q[1] = *(const u32x4*)(pa + k0 + c0 + 16);
    b.q[0] = *(const u32x4*)(pb + k0 + bk0);
    b.q[1] = *(const u32x4*)(pb + k0 + bk0 + 16);
    acc = wmma_bf16(a, b, acc);
  }
  const int col = n0 + l15;
  const int mbase = (lane >> 4) * 8;
#pragma unroll
  for (int r = 0; r < 8; ++r) {
    const int m = mbase + r;
    if (m < MB) {
      const float v = acc[r] + Xh[(size_t)(t * 4 + m) * MD + col];
      const float hh = tanhf(v);
      const float z = zbuf[m * MD + col];
      const float ho = hcur[m * MD + col];
      const float hn = (1.f - z) * ho + z * hh;
      hcur[m * MD + col] = hn;
      const unsigned short hb = f2bf(hn);
      hbf[m * MD + col] = hb;
      h_all[(size_t)(t * 4 + m) * MD + col] = hn;
      hbf_all[(size_t)(t * 4 + m) * MD + col] = hb;
    }
  }
}

// zero the GRU state (hcur f32[4D], zbuf f32[4D], hbf/rhbf u16[16D])
__global__ __launch_bounds__(256) void gru_init_kernel(
    float* hcur, float* zbuf, unsigned short* hbf, unsigned short* rhbf) {
  int i = blockIdx.x * 256 + threadIdx.x;
  if (i < 4 * MD) { hcur[i] = 0.f; zbuf[i] = 0.f; }
  if (i < 16 * MD) { hbf[i] = 0; rhbf[i] = 0; }
}

// =====================================================================
// 7) Softmax over memory slots: scores [T, NS] f32 -> attn_bf [T, NS] bf16
// =====================================================================
__global__ __launch_bounds__(256) void softmax_kernel(
    const float* __restrict__ scores, unsigned short* __restrict__ attn) {
  const int t = blockIdx.x;
  const int tid = threadIdx.x;
  const float scale = 0.03125f;   // 1/sqrt(1024)
  __shared__ float red[256];
  float v0 = scores[(size_t)t * MNS + tid] * scale;
  float v1 = scores[(size_t)t * MNS + tid + 256] * scale;
  float lmax = fmaxf(v0, v1);
  red[tid] = lmax; __syncthreads();
  for (int off = 128; off > 0; off >>= 1) {
    if (tid < off) red[tid] = fmaxf(red[tid], red[tid + off]);
    __syncthreads();
  }
  const float mx = red[0]; __syncthreads();
  const float e0 = expf(v0 - mx), e1 = expf(v1 - mx);
  red[tid] = e0 + e1; __syncthreads();
  for (int off = 128; off > 0; off >>= 1) {
    if (tid < off) red[tid] += red[tid + off];
    __syncthreads();
  }
  const float inv = 1.f / red[0];
  attn[(size_t)t * MNS + tid]       = f2bf(e0 * inv);
  attn[(size_t)t * MNS + tid + 256] = f2bf(e1 * inv);
}

// =====================================================================
// 8) IGPM elementwise: igpm = bcm * sigmoid(G) + P
// =====================================================================
__global__ __launch_bounds__(256) void igpm_kernel(
    const float* __restrict__ bcm, const float* __restrict__ G,
    const float* __restrict__ P, float* __restrict__ out, int n) {
  int i = blockIdx.x * 256 + threadIdx.x;
  if (i < n) out[i] = bcm[i] * sigmoidf_(G[i]) + P[i];
}

// =====================================================================
// Host orchestration
// =====================================================================
static inline size_t align_up(size_t x, size_t a) { return (x + a - 1) & ~(a - 1); }

extern "C" void kernel_launch(void* const* d_in, const int* in_sizes, int n_in,
                              void* d_out, int out_size, void* d_ws, size_t ws_size,
                              hipStream_t stream) {
  (void)in_sizes; (void)n_in; (void)out_size; (void)ws_size;
  const int*   ids     = (const int*)  d_in[0];
  const float* tok_emb = (const float*)d_in[1];
  const float* pos_emb = (const float*)d_in[2];
  const float* ln_in_g = (const float*)d_in[3];
  const float* ln_in_b = (const float*)d_in[4];
  const float* Wz      = (const float*)d_in[5];
  const float* bz      = (const float*)d_in[6];
  const float* Wr      = (const float*)d_in[7];
  const float* br      = (const float*)d_in[8];
  const float* Wh      = (const float*)d_in[9];
  const float* bh      = (const float*)d_in[10];
  const float* mem     = (const float*)d_in[11];
  const float* Wg      = (const float*)d_in[12];
  const float* bg      = (const float*)d_in[13];
  const float* Wp      = (const float*)d_in[14];
  const float* bp      = (const float*)d_in[15];
  const float* ln_o_g  = (const float*)d_in[16];
  const float* ln_o_b  = (const float*)d_in[17];
  const float* Wo      = (const float*)d_in[18];
  const float* bo      = (const float*)d_in[19];
  float* out = (float*)d_out;

  // -------- workspace carve-up --------
  char* p = (char*)d_ws;
  size_t off = 0;
  auto alloc = [&](size_t bytes) { void* r = p + off; off = align_up(off + bytes, 256); return r; };

  unsigned short* x_bf    = (unsigned short*)alloc((size_t)MT * MD * 2);
  float* Xz               = (float*)alloc((size_t)MT * MD * 4);
  float* Xr               = (float*)alloc((size_t)MT * MD * 4);
  float* Xh               = (float*)alloc((size_t)MT * MD * 4);
  float* h_all            = (float*)alloc((size_t)MT * MD * 4);
  unsigned short* hbf_all = (unsigned short*)alloc((size_t)MT * MD * 2);
  float* hcur             = (float*)alloc((size_t)4 * MD * 4);
  float* zbuf             = (float*)alloc((size_t)4 * MD * 4);
  unsigned short* hbf     = (unsigned short*)alloc((size_t)16 * MD * 2);
  unsigned short* rhbf    = (unsigned short*)alloc((size_t)16 * MD * 2);
  float* scores           = (float*)alloc((size_t)MT * MNS * 4);
  unsigned short* attn_bf = (unsigned short*)alloc((size_t)MT * MNS * 2);
  float* bcm              = (float*)alloc((size_t)MT * MD * 4);
  unsigned short* bcm_bf  = (unsigned short*)alloc((size_t)MT * MD * 2);
  float* Gbuf             = (float*)alloc((size_t)MT * MD * 4);
  float* Pbuf             = (float*)alloc((size_t)MT * MD * 4);
  float* igpm             = (float*)alloc((size_t)MT * MD * 4);
  unsigned short* fin_bf  = (unsigned short*)alloc((size_t)MT * MD * 2);
  unsigned short* mem_bf  = (unsigned short*)alloc((size_t)MNS * MD * 2);
  unsigned short* memT_bf = (unsigned short*)alloc((size_t)MD * MNS * 2);
  unsigned short* Wzx_t   = (unsigned short*)alloc((size_t)MD * MD * 2);
  unsigned short* Wzh_t   = (unsigned short*)alloc((size_t)MD * MD * 2);
  unsigned short* Wrx_t   = (unsigned short*)alloc((size_t)MD * MD * 2);
  unsigned short* Wrh_t   = (unsigned short*)alloc((size_t)MD * MD * 2);
  unsigned short* Whx_t   = (unsigned short*)alloc((size_t)MD * MD * 2);
  unsigned short* Whh_t   = (unsigned short*)alloc((size_t)MD * MD * 2);
  unsigned short* Wg_t    = (unsigned short*)alloc((size_t)MD * MD * 2);
  unsigned short* Wp_t    = (unsigned short*)alloc((size_t)MD * MD * 2);
  unsigned short* Wo_t    = (unsigned short*)alloc((size_t)MV * MD * 2);

  // -------- stage 0: weight conversion / transposition (bf16, [N,K]) --------
  hipLaunchKernelGGL(gru_init_kernel, dim3((16 * MD + 255) / 256), dim3(256), 0, stream,
                     hcur, zbuf, hbf, rhbf);
  hipLaunchKernelGGL(convert_bf16_kernel, dim3((MNS * MD + 255) / 256), dim3(256), 0, stream,
                     mem, mem_bf, MNS * MD);
  hipLaunchKernelGGL(transpose_cvt_kernel, dim3(MD / 32, MNS / 32), dim3(256), 0, stream,
                     mem, memT_bf, MNS, MD);
  hipLaunchKernelGGL(transpose_cvt_kernel, dim3(MD / 32, MD / 32), dim3(256), 0, stream,
                     Wz, Wzx_t, MD, MD);
  hipLaunchKernelGGL(transpose_cvt_kernel, dim3(MD / 32, MD / 32), dim3(256), 0, stream,
                     Wz + (size_t)MD * MD, Wzh_t, MD, MD);
  hipLaunchKernelGGL(transpose_cvt_kernel, dim3(MD / 32, MD / 32), dim3(256), 0, stream,
                     Wr, Wrx_t, MD, MD);
  hipLaunchKernelGGL(transpose_cvt_kernel, dim3(MD / 32, MD / 32), dim3(256), 0, stream,
                     Wr + (size_t)MD * MD, Wrh_t, MD, MD);
  hipLaunchKernelGGL(transpose_cvt_kernel, dim3(MD / 32, MD / 32), dim3(256), 0, stream,
                     Wh, Whx_t, MD, MD);
  hipLaunchKernelGGL(transpose_cvt_kernel, dim3(MD / 32, MD / 32), dim3(256), 0, stream,
                     Wh + (size_t)MD * MD, Whh_t, MD, MD);
  hipLaunchKernelGGL(transpose_cvt_kernel, dim3(MD / 32, MD / 32), dim3(256), 0, stream,
                     Wg, Wg_t, MD, MD);
  hipLaunchKernelGGL(transpose_cvt_kernel, dim3(MD / 32, MD / 32), dim3(256), 0, stream,
                     Wp, Wp_t, MD, MD);
  hipLaunchKernelGGL(transpose_cvt_kernel, dim3(MV / 32, MD / 32), dim3(256), 0, stream,
                     Wo, Wo_t, MD, MV);

  // -------- stage 1: embed + LN --------
  hipLaunchKernelGGL(embed_ln_kernel, dim3(MT), dim3(256), 0, stream,
                     ids, tok_emb, pos_emb, ln_in_g, ln_in_b, x_bf);

  // -------- stage 2: hoisted input-half gate GEMMs (token-parallel) --------
  hipLaunchKernelGGL(gemm_bf16_kernel, dim3(MD / BN, MT / BM), dim3(256), 0, stream,
                     x_bf, Wzx_t, bz, (const float*)nullptr, Xz, MT, MD, MD, 0);
  hipLaunchKernelGGL(gemm_bf16_kernel, dim3(MD / BN, MT / BM), dim3(256), 0, stream,
                     x_bf, Wrx_t, br, (const float*)nullptr, Xr, MT, MD, MD, 0);
  hipLaunchKernelGGL(gemm_bf16_kernel, dim3(MD / BN, MT / BM), dim3(256), 0, stream,
                     x_bf, Whx_t, bh, (const float*)nullptr, Xh, MT, MD, MD, 0);

  // -------- stage 3: sequential GRU recurrence (weights stay in L2) --------
  for (int t = 0; t < MS; ++t) {
    hipLaunchKernelGGL(gru_zr_kernel, dim3(MD / 64, 2), dim3(128), 0, stream,
                       hbf, hcur, Wzh_t, Wrh_t, Xz, Xr, zbuf, rhbf, t);
    hipLaunchKernelGGL(gru_h_kernel, dim3(MD / 64, 1), dim3(128), 0, stream,
                       rhbf, Whh_t, Xh, zbuf, hcur, hbf, h_all, hbf_all, t);
  }

  // -------- stage 4: BCM memory read --------
  hipLaunchKernelGGL(gemm_bf16_kernel, dim3(MNS / BN, MT / BM), dim3(256), 0, stream,
                     hbf_all, mem_bf, (const float*)nullptr, (const float*)nullptr,
                     scores, MT, MNS, MD, 0);
  hipLaunchKernelGGL(softmax_kernel, dim3(MT), dim3(256), 0, stream, scores, attn_bf);
  hipLaunchKernelGGL(gemm_bf16_kernel, dim3(MD / BN, MT / BM), dim3(256), 0, stream,
                     attn_bf, memT_bf, (const float*)nullptr, h_all, bcm, MT, MD, MNS, 0);

  // -------- stage 5: IGPM --------
  hipLaunchKernelGGL(convert_bf16_kernel, dim3((MT * MD + 255) / 256), dim3(256), 0, stream,
                     bcm, bcm_bf, MT * MD);
  hipLaunchKernelGGL(gemm_bf16_kernel, dim3(MD / BN, MT / BM), dim3(256), 0, stream,
                     bcm_bf, Wg_t, bg, (const float*)nullptr, Gbuf, MT, MD, MD, 0);
  hipLaunchKernelGGL(gemm_bf16_kernel, dim3(MD / BN, MT / BM), dim3(256), 0, stream,
                     bcm_bf, Wp_t, bp, (const float*)nullptr, Pbuf, MT, MD, MD, 0);
  hipLaunchKernelGGL(igpm_kernel, dim3((MT * MD + 255) / 256), dim3(256), 0, stream,
                     bcm, Gbuf, Pbuf, igpm, MT * MD);

  // -------- stage 6: output LN + vocab projection --------
  hipLaunchKernelGGL(ln_out_kernel, dim3(MT), dim3(256), 0, stream,
                     igpm, ln_o_g, ln_o_b, fin_bf);
  hipLaunchKernelGGL(gemm_bf16_kernel, dim3(MV / BN, MT / BM), dim3(256), 0, stream,
                     fin_bf, Wo_t, bo, (const float*)nullptr, out, MT, MV, MD, 1);
}
